// Gate_7241314861587
// MI455X (gfx1250) — matrix operand order repaired and checked by reading, not compiled, not run on hardware
//
#include <hip/hip_runtime.h>
#include <math.h>

// MoE router gate: logits = x @ W^T, sigmoid, top-8/256, normalize.
// x: [8192,4096] f32, W: [256,4096] f32.
// d_out = gates [8192*8] f32 followed by indices [8192*8] i32.

#define TOKENS 8192
#define DIM    4096
#define NEXP   256
#define TOPK   8

#define M_TILE 64          // tokens per block
#define KSTAGE 64          // K staged to LDS per barrier (2 WMMA k-steps)
#define WAVES  8
#define BLOCK  (WAVES * 32)
#define SA_STRIDE 72       // 64 + 8 pad; 144B row stride keeps 16B alignment

typedef __attribute__((ext_vector_type(16))) __bf16 v16bf;
typedef __attribute__((ext_vector_type(8)))  __bf16 v8bf;
typedef __attribute__((ext_vector_type(8)))  float  v8f;

static __device__ __forceinline__ void cvt_f4(v16bf& v, int base, float4 f) {
    v[base + 0] = (__bf16)f.x;
    v[base + 1] = (__bf16)f.y;
    v[base + 2] = (__bf16)f.z;
    v[base + 3] = (__bf16)f.w;
}

__global__ __launch_bounds__(BLOCK)
void moe_gate_kernel(const float* __restrict__ x,
                     const float* __restrict__ w,
                     float* __restrict__ gates_out,
                     int*   __restrict__ idx_out)
{
    __shared__ __bf16 sA[M_TILE][SA_STRIDE];   // staged x chunk (bf16), ~9.2KB
    __shared__ float  sS[32][NEXP];            // sigmoid scores, 32 tokens/pass

    const int tid   = threadIdx.x;
    const int wave  = tid >> 5;
    const int lane  = tid & 31;
    const int mBase = blockIdx.x * M_TILE;

    // ---- B (weight) fragment addressing: wave owns experts [wave*32, wave*32+32)
    // B k order used here: lanes 0-15 -> K=0..15 of column N=lane,
    //                        lanes 16-31 -> K=16..31 of column N=lane-16.
    const int   eloc  = lane & 15;
    const int   kloB  = (lane < 16) ? 0 : 16;
    const float* wrow0 = w + (size_t)(wave * 32 + eloc)      * DIM + kloB;
    const float* wrow1 = w + (size_t)(wave * 32 + 16 + eloc) * DIM + kloB;

    // ---- A fragment addressing
    // A k order used here: lanes 0-15 -> row M=lane,   K in {0..15}
    //                        lanes 16-31-> row M=lane-16,K in {16..31}
    const int mA    = lane & 15;
    const int koffA = (lane < 16) ? 0 : 16;

    v8f acc[4][2];
#pragma unroll
    for (int mi = 0; mi < 4; ++mi)
#pragma unroll
        for (int t = 0; t < 2; ++t)
            acc[mi][t] = (v8f)0.0f;

    // ---- x staging coords: 64x64 floats per stage, 16 per thread (quarter-row)
    const int srow = tid >> 2;
    const int scol = (tid & 3) * 16;
    const float* xrow = x + (size_t)(mBase + srow) * DIM + scol;

    for (int k0 = 0; k0 < DIM; k0 += KSTAGE) {
        __syncthreads();
        // stage x chunk -> LDS as bf16 (16 floats -> two 16B LDS stores)
        {
            float4 f0 = *(const float4*)(xrow + k0);
            float4 f1 = *(const float4*)(xrow + k0 + 4);
            float4 f2 = *(const float4*)(xrow + k0 + 8);
            float4 f3 = *(const float4*)(xrow + k0 + 12);
            v8bf lo, hi;
            lo[0] = (__bf16)f0.x; lo[1] = (__bf16)f0.y;
            lo[2] = (__bf16)f0.z; lo[3] = (__bf16)f0.w;
            lo[4] = (__bf16)f1.x; lo[5] = (__bf16)f1.y;
            lo[6] = (__bf16)f1.z; lo[7] = (__bf16)f1.w;
            hi[0] = (__bf16)f2.x; hi[1] = (__bf16)f2.y;
            hi[2] = (__bf16)f2.z; hi[3] = (__bf16)f2.w;
            hi[4] = (__bf16)f3.x; hi[5] = (__bf16)f3.y;
            hi[6] = (__bf16)f3.z; hi[7] = (__bf16)f3.w;
            *(v8bf*)&sA[srow][scol]     = lo;
            *(v8bf*)&sA[srow][scol + 8] = hi;
        }
        __syncthreads();

        if (k0 + KSTAGE < DIM) {                   // global_prefetch_b8 next stage
            __builtin_prefetch(xrow  + k0 + KSTAGE, 0, 3);
            __builtin_prefetch(wrow0 + k0 + KSTAGE, 0, 3);
            __builtin_prefetch(wrow1 + k0 + KSTAGE, 0, 3);
        }

#pragma unroll
        for (int ks = 0; ks < KSTAGE; ks += 32) {
            // B fragments: 16 contiguous k values per lane, convert f32->bf16
            v16bf b0, b1;
            cvt_f4(b0, 0,  *(const float4*)(wrow0 + k0 + ks));
            cvt_f4(b0, 4,  *(const float4*)(wrow0 + k0 + ks + 4));
            cvt_f4(b0, 8,  *(const float4*)(wrow0 + k0 + ks + 8));
            cvt_f4(b0, 12, *(const float4*)(wrow0 + k0 + ks + 12));
            cvt_f4(b1, 0,  *(const float4*)(wrow1 + k0 + ks));
            cvt_f4(b1, 4,  *(const float4*)(wrow1 + k0 + ks + 4));
            cvt_f4(b1, 8,  *(const float4*)(wrow1 + k0 + ks + 8));
            cvt_f4(b1, 12, *(const float4*)(wrow1 + k0 + ks + 12));

            // Load ALL four A fragments first (distinct regs) so the 8
            // ds_load_b128 clause up and the 8 WMMAs issue back-to-back.
            v16bf afr[4];
#pragma unroll
            for (int mi = 0; mi < 4; ++mi) {
                const __bf16* ap = &sA[mi * 16 + mA][ks];
                v8bf alo = *(const v8bf*)(ap + koffA);        // K block 0
                v8bf ahi = *(const v8bf*)(ap + 8 + koffA);    // K block 1
#pragma unroll
                for (int j = 0; j < 8; ++j) {
                    afr[mi][j]     = alo[j];
                    afr[mi][j + 8] = ahi[j];
                }
            }
#pragma unroll
            for (int mi = 0; mi < 4; ++mi) {
                acc[mi][0] = __builtin_amdgcn_wmma_f32_16x16x32_bf16(
                    false, afr[mi], false, b0, (short)0, acc[mi][0], false, false);
                acc[mi][1] = __builtin_amdgcn_wmma_f32_16x16x32_bf16(
                    false, afr[mi], false, b1, (short)0, acc[mi][1], false, false);
            }
        }
    }

    // ---- epilogue: two passes of 32 tokens (keeps LDS ~41KB).
    // Outer loops intentionally NOT unrolled (code-size control); inner
    // round/lane loops stay unrolled so tv/ti/sv remain register-resident.
    for (int p = 0; p < 2; ++p) {
        __syncthreads();
        // scatter sigmoid(scores) for tokens [p*32, p*32+32)
        for (int half = 0; half < 2; ++half) {
            const int mi = 2 * p + half;
#pragma unroll
            for (int t = 0; t < 2; ++t) {
#pragma unroll
                for (int r = 0; r < 8; ++r) {
                    // C layout: VGPR r, lane<16 -> M=r,N=lane; lane>=16 -> M=r+8,N=lane-16
                    int mrow = half * 16 + r + ((lane < 16) ? 0 : 8);
                    int e    = wave * 32 + t * 16 + (lane & 15);
                    float v  = acc[mi][t][r];
                    sS[mrow][e] = 1.0f / (1.0f + __expf(-v));
                }
            }
        }
        __syncthreads();

        // top-8: each wave handles 4 tokens of this pass
        for (int tt = 0; tt < 4; ++tt) {
            const int rloc = wave * 4 + tt;          // row within sS
            const int gtok = mBase + p * 32 + rloc;  // global token

            float sv[8];
#pragma unroll
            for (int j = 0; j < 8; ++j) sv[j] = sS[rloc][lane * 8 + j];

            float tv[TOPK];
            int   ti[TOPK];
            float sum = 0.0f;
#pragma unroll
            for (int r = 0; r < TOPK; ++r) {
                float bv = sv[0];
                int   bj = 0;
#pragma unroll
                for (int j = 1; j < 8; ++j)
                    if (sv[j] > bv) { bv = sv[j]; bj = j; }
                int bi = lane * 8 + bj;
                // wave32 argmax reduction, tie-break on lower index (match top_k)
#pragma unroll
                for (int off = 16; off >= 1; off >>= 1) {
                    float ov = __shfl_xor(bv, off);
                    int   oi = __shfl_xor(bi, off);
                    if (ov > bv || (ov == bv && oi < bi)) { bv = ov; bi = oi; }
                }
                tv[r] = bv; ti[r] = bi; sum += bv;
#pragma unroll
                for (int j = 0; j < 8; ++j)
                    if (bi == lane * 8 + j) sv[j] = -INFINITY;
            }

            if (lane == 0) {
                float inv = 1.0f / sum;
#pragma unroll
                for (int r = 0; r < TOPK; ++r) {
                    gates_out[(size_t)gtok * TOPK + r] = tv[r] * inv;
                    idx_out  [(size_t)gtok * TOPK + r] = ti[r];
                }
            }
        }
    }
}

extern "C" void kernel_launch(void* const* d_in, const int* in_sizes, int n_in,
                              void* d_out, int out_size, void* d_ws, size_t ws_size,
                              hipStream_t stream) {
    (void)in_sizes; (void)n_in; (void)out_size; (void)d_ws; (void)ws_size;
    const float* x = (const float*)d_in[0];
    const float* w = (const float*)d_in[1];
    float* gates = (float*)d_out;
    int*   idx   = (int*)((float*)d_out + (size_t)TOKENS * TOPK);

    dim3 grid(TOKENS / M_TILE);   // 128 blocks
    dim3 block(BLOCK);            // 256 threads = 8 wave32
    moe_gate_kernel<<<grid, block, 0, stream>>>(x, w, gates, idx);
}
